// TimeSampled_62139586838597
// MI455X (gfx1250) — matrix operand order repaired and checked by reading, not compiled, not run on hardware
//
#include <hip/hip_runtime.h>
#include <stdint.h>

#define HID    16
#define CIN    17       // 1 input ch + 16 hidden
#define GATES  64
#define NQ     6        // 5 chunks of (2 taps x 16ch) + 1 chunk for channel 16
#define IMG    256
#define TFULL  110
#define SEQ    30
#define TILE_W 32
#define TILE_H 4
#define PIX    128      // pixels per block (32x4)
#define HALO_W 34
#define HALO_H 6
#define NPOS   (HALO_H * HALO_W)   // 204 halo positions

typedef __attribute__((ext_vector_type(16))) __bf16         v16bf;
typedef __attribute__((ext_vector_type(16))) unsigned short v16u;
typedef __attribute__((ext_vector_type(8)))  float          v8f;

__device__ __forceinline__ unsigned short f2bf(float f) {
  union { float f; unsigned u; } x; x.f = f;
  unsigned u = x.u;
  return (unsigned short)((u + 0x7fffu + ((u >> 16) & 1u)) >> 16);  // RNE
}
__device__ __forceinline__ float sigm(float x) {        // 1/(1+e^-x), one v_exp
  return 1.f / (1.f + __expf(-x));
}
__device__ __forceinline__ float tanh_fast(float x) {   // (e^2x-1)/(e^2x+1)
  float t = __expf(2.f * x);
  return (t - 1.f) / (t + 1.f);
}

// ---------------- zero workspace ----------------
__global__ void ws_zero_kernel(float* p, size_t n) {
  size_t i = (size_t)blockIdx.x * blockDim.x + threadIdx.x;
  size_t st = (size_t)gridDim.x * blockDim.x;
  for (; i < n; i += st) p[i] = 0.f;
}

// ------- pack conv weights into bf16 WMMA A-fragment layout -------
// K-slot order: chunk q<5, slot s: tap = 2q + (s>>4), ch = s&15 (tap 9 -> 0).
//               chunk 5,  slot s: (tap = s, ch = 16) for s<9, else 0.
// A 16x32 fragment half j -> slot: lane<16: j;
//                              lane>=16: j+16.
// Stored [m_tile(4)][q(6)][lane(32)][half(16)]: one 32B load per lane.
__global__ void pack_w_kernel(const float* __restrict__ conv_w,
                              unsigned short* __restrict__ wbf) {
  int i = blockIdx.x * blockDim.x + threadIdx.x;
  if (i >= 4 * NQ * 32 * 16) return;
  int j    = i & 15;
  int lane = (i >> 4) & 31;
  int mq   = i >> 9;
  int q    = mq % NQ;
  int mt   = mq / NQ;
  int m    = lane & 15;
  int s = (lane < 16) ? j : j + 16;
  float v = 0.f;
  int g = mt * 16 + m;
  if (q < 5) {
    int tap = 2 * q + (s >> 4), c = s & 15;
    if (tap < 9) v = conv_w[(g * CIN + c) * 9 + tap];
  } else {
    if (s < 9) v = conv_w[(g * CIN + 16) * 9 + s];
  }
  wbf[i] = f2bf(v);
}

// ---------------- fused ConvLSTM step ----------------
// h state is stored bf16 in global (it feeds WMMA as bf16 anyway); c stays f32.
__global__ __launch_bounds__(256) void convlstm_step_kernel(
    const float* __restrict__ xin, const unsigned short* __restrict__ wbf,
    const float* __restrict__ bias, const unsigned short* __restrict__ hin,
    unsigned short* __restrict__ hout, float* __restrict__ cbuf,
    float* __restrict__ pooled, int t, int t_next, int s) {
  // z channels 0..15 = [x_t, h0..h14], channel-minor groups of 16 so a
  // B-fragment half-chunk (one tap x 16ch) is one contiguous 32B LDS load.
  __shared__ __align__(32) unsigned short zlds[NPOS * 16];       // 6528 B
  __shared__ __align__(32) unsigned short c16h[NPOS + 4];        // h[15] halo plane
  __shared__ __align__(32) unsigned short c16pix[2 * PIX * 16];  // [sel][p][16]; sel1 = zeros
  __shared__ float glds[GATES * PIX];
  __shared__ float rlds[HID * 4];

  int tid = threadIdx.x;
  int bid = blockIdx.x;
  int bb  = bid >> 9;                 // batch
  int rem = bid & 511;
  int y0  = (rem >> 3) * TILE_H;
  int x0  = (rem & 7) * TILE_W;

  // Prefetch next step's x-frame tile (global_prefetch_b8).
  if (tid < TILE_H) {
    const float* nxt = xin + ((size_t)(bb * TFULL + t_next) * IMG + (y0 + tid)) * IMG + x0;
    __builtin_prefetch(nxt, 0, 1);
  }

  // ---- Phase 1: stage halo (bf16) + zero c16pix ----
  const float* xframe = xin + (size_t)(bb * TFULL + t) * IMG * IMG;
  const unsigned short* hbase = hin + (size_t)bb * HID * IMG * IMG;
  for (int e = tid; e < CIN * NPOS; e += 256) {
    int c   = e / NPOS;          // 0..16 ; plane-major for coalesced global reads
    int pos = e % NPOS;
    int ly = pos / HALO_W, lx = pos % HALO_W;
    int y = y0 + ly - 1, x = x0 + lx - 1;
    unsigned short bv = 0;
    if ((unsigned)y < IMG && (unsigned)x < IMG)
      bv = (c == 0) ? f2bf(xframe[y * IMG + x])
                    : hbase[((size_t)(c - 1) * IMG + y) * IMG + x];
    if (c < 16) zlds[pos * 16 + c] = bv;
    else        c16h[pos] = bv;
  }
  for (int e = tid; e < 2 * PIX * 16; e += 256) c16pix[e] = 0;
  __syncthreads();

  // ---- Phase 2: expand channel-16 per-pixel tap fragments ----
  for (int e = tid; e < PIX * 9; e += 256) {
    int p = e / 9, j = e % 9;
    int py = p >> 5, px = p & 31;
    int dy = j / 3, dx = j - dy * 3;
    c16pix[p * 16 + j] = c16h[(py + dy) * HALO_W + (px + dx)];
  }
  __syncthreads();

  // ---- Implicit GEMM: 8 waves; wave -> (m_tile = w&3, n_group = w>>2) ----
  int wave = tid >> 5, lane = tid & 31;
  int mt = wave & 3, ng = wave >> 2;
  int lohi = (lane >> 4) & 1;          // 0: K 0..15 (even tap), 1: K 16..31 (odd tap)

  v16u afr[NQ];
#pragma unroll
  for (int q = 0; q < NQ; ++q)
    afr[q] = *(const v16u*)(wbf + (((mt * NQ + q) * 32 + lane) << 4));

  int n = lane & 15;
  const unsigned short* czero = &c16pix[PIX * 16];  // 32B of zeros (tap 9)

#pragma unroll
  for (int i = 0; i < 4; ++i) {
    int nt = ng * 4 + i;
    int p  = nt * 16 + n;
    int py = p >> 5, px = p & 31;
    v8f acc = {0.f, 0.f, 0.f, 0.f, 0.f, 0.f, 0.f, 0.f};
#pragma unroll
    for (int q = 0; q < 5; ++q) {
      const int t0 = 2 * q, t1 = 2 * q + 1;           // compile-time taps
      const unsigned short* lo = &zlds[((py + t0 / 3) * HALO_W + (px + t0 % 3)) * 16];
      const unsigned short* hi = (t1 < 9)
          ? &zlds[((py + t1 / 3) * HALO_W + (px + t1 % 3)) * 16]
          : czero;
      v16u bu = *(const v16u*)(lohi ? hi : lo);       // branchless select, 32B load
      acc = __builtin_amdgcn_wmma_f32_16x16x32_bf16(
          false, __builtin_bit_cast(v16bf, afr[q]),
          false, __builtin_bit_cast(v16bf, bu),
          (short)0, acc, false, false);
    }
    {
      v16u bu = *(const v16u*)&c16pix[(lohi * PIX + p) * 16];  // sel1 = zeros
      acc = __builtin_amdgcn_wmma_f32_16x16x32_bf16(
          false, __builtin_bit_cast(v16bf, afr[5]),
          false, __builtin_bit_cast(v16bf, bu),
          (short)0, acc, false, false);
    }
    // D layout: vgpr r, lane<16 -> M=r,N=lane ; lane>=16 -> M=r+8,N=lane-16
#pragma unroll
    for (int r = 0; r < 8; ++r)
      glds[(mt * 16 + r + lohi * 8) * PIX + p] = acc[r];
  }
  __syncthreads();

  // ---- LSTM pointwise epilogue, coalesced: lane <-> pixel x, channel loop ----
  // thread -> pixel p = tid&127 (consecutive lanes = consecutive x), 8 channels each.
  {
    int p  = tid & 127;
    int py = p >> 5, px = p & 31;
    int y = y0 + py, x = x0 + px;
    int chbase = (tid >> 7) * 8;     // threads 0-127: ch 0..7 ; 128-255: ch 8..15
    unsigned short* hb = hout + (size_t)bb * HID * IMG * IMG;
    float*          cb = cbuf + (size_t)bb * HID * IMG * IMG;
#pragma unroll
    for (int j = 0; j < 8; ++j) {
      int ch = chbase + j;
      size_t idx = ((size_t)ch * IMG + y) * IMG + x;
      float gi = glds[ch * PIX + p]        + bias[ch];
      float gf = glds[(16 + ch) * PIX + p] + bias[16 + ch];
      float go = glds[(32 + ch) * PIX + p] + bias[32 + ch];
      float gg = glds[(48 + ch) * PIX + p] + bias[48 + ch];
      float cn = sigm(gf) * cb[idx] + sigm(gi) * tanh_fast(gg);
      float hn = sigm(go) * tanh_fast(cn);
      cb[idx] = cn;                       // coalesced b32 store
      hb[idx] = f2bf(hn);                 // coalesced b16 store
      // ch is wave-uniform here: wave32 shuffle reduction over 32 pixels
      float r = hn;
#pragma unroll
      for (int off = 16; off > 0; off >>= 1) r += __shfl_xor(r, off, 32);
      if (lane == 0) rlds[ch * 4 + (wave & 3)] = r;
    }
  }
  __syncthreads();
  if (tid < HID) {
    float tot = rlds[tid * 4] + rlds[tid * 4 + 1] + rlds[tid * 4 + 2] + rlds[tid * 4 + 3];
    atomicAdd(&pooled[(bb * HID + tid) * SEQ + s], tot);
  }
}

// ---------------- final FC: (B, 480) @ fc_w^T + fc_b ----------------
__global__ void fc_kernel(const float* __restrict__ pooled,
                          const float* __restrict__ fcw,
                          const float* __restrict__ fcb, float* __restrict__ out) {
  __shared__ float red[6 * 32];
  int tid = threadIdx.x;         // 192 threads = 6 outputs x 32 lanes
  int o = tid >> 5, lane = tid & 31;
  int b = o / 3, j = o % 3;
  float p = 0.f;
  for (int k = lane; k < HID * SEQ; k += 32)
    p += fcw[j * (HID * SEQ) + k] * pooled[b * (HID * SEQ) + k];
  red[o * 32 + lane] = p;
  __syncthreads();
  if (lane == 0) {
    float sacc = 0.f;
    for (int q = 0; q < 32; ++q) sacc += red[o * 32 + q];
    out[b * 3 + j] = sacc * (1.f / (IMG * IMG)) + fcb[j];  // spatial mean folded in
  }
}

extern "C" void kernel_launch(void* const* d_in, const int* in_sizes, int n_in,
                              void* d_out, int out_size, void* d_ws, size_t ws_size,
                              hipStream_t stream) {
  (void)in_sizes; (void)n_in; (void)out_size; (void)ws_size;
  const float* x      = (const float*)d_in[0];
  const float* conv_w = (const float*)d_in[1];
  const float* conv_b = (const float*)d_in[2];
  const float* fc_w   = (const float*)d_in[3];
  const float* fc_b   = (const float*)d_in[4];
  float* out = (float*)d_out;

  // Workspace (~16.8 MB): packed bf16 weights | h0,h1 (bf16) | c (f32) | pooled
  char* ws = (char*)d_ws;
  unsigned short* wbf = (unsigned short*)ws;          // 24576 B used, 32768 reserved
  const size_t PLANE = (size_t)2 * HID * IMG * IMG;   // 2,097,152 elements
  unsigned short* h0 = (unsigned short*)(ws + 32768);
  unsigned short* h1 = h0 + PLANE;                    // bf16, 4 MB each
  float* cb     = (float*)(ws + 32768 + 2 * PLANE * sizeof(unsigned short));
  float* pooled = cb + PLANE;

  // Zero h0,h1,c,pooled (contiguous; bf16 zeros == f32 zeros bytewise).
  size_t nz = (2 * PLANE * sizeof(unsigned short) + PLANE * sizeof(float)) / 4
              + (size_t)2 * HID * SEQ;
  ws_zero_kernel<<<2048, 256, 0, stream>>>((float*)h0, nz);
  pack_w_kernel<<<(4 * NQ * 32 * 16 + 255) / 256, 256, 0, stream>>>(conv_w, wbf);

  static const int IDX[SEQ] = {0, 10, 11, 12, 13, 14, 15, 16, 17, 18, 19,
                               20, 21, 22, 23, 24, 25, 26, 27, 28, 29,
                               29, 39, 49, 59, 69, 79, 89, 99, 109};

  const unsigned short* hin = h0;
  unsigned short* hout = h1;
  for (int s = 0; s < SEQ; ++s) {
    int tn = IDX[(s < SEQ - 1) ? s + 1 : s];
    convlstm_step_kernel<<<1024, 256, 0, stream>>>(x, wbf, conv_b, hin, hout,
                                                   cb, pooled, IDX[s], tn, s);
    unsigned short* tmp = hout; hout = (unsigned short*)hin; hin = tmp;
  }
  fc_kernel<<<1, 192, 0, stream>>>(pooled, fc_w, fc_b, out);
}
